// MoELayer_85014582657080
// MI455X (gfx1250) — compile-verified, matrix operand order checked by reading, not run on hardware
//
#include <hip/hip_runtime.h>

// ---------------- problem constants (from reference) ----------------
#define T_ 2048
#define H_ 2048
#define I_ 1408
#define E_ 8
#define K_ 2
#define NPAIR  (T_ * K_)            // 4096 (token, expert-choice) pairs
#define PSLOTS (NPAIR + E_ * 64)    // per-expert regions padded to 64 rows
#define MTILES (PSLOTS / 64)        // 72 max 64-row M blocks

typedef __bf16 bf16;
typedef __bf16 v16bf __attribute__((ext_vector_type(16)));
typedef __bf16 v8bf  __attribute__((ext_vector_type(8)));
typedef __bf16 v4bf  __attribute__((ext_vector_type(4)));
typedef float  v8f   __attribute__((ext_vector_type(8)));

// ---------------- workspace layout (bytes) ----------------
#define SZ_W13B  ((size_t)E_ * 2 * I_ * H_ * 2)   //  92,274,688
#define SZ_W2B   ((size_t)E_ * H_ * I_ * 2)       //  46,137,344
#define SZ_HIDB  ((size_t)T_ * H_ * 2)            //   8,388,608
#define SZ_ACTB  ((size_t)PSLOTS * I_ * 2)        //  12,976,128
#define SZ_OUTP  ((size_t)PSLOTS * H_ * 4)        //  37,748,736
#define OFF_W13B 0ull
#define OFF_W2B  (OFF_W13B + SZ_W13B)
#define OFF_HIDB (OFF_W2B  + SZ_W2B)
#define OFF_ACTB (OFF_HIDB + SZ_HIDB)
#define OFF_OUTP (OFF_ACTB + SZ_ACTB)
#define OFF_META (OFF_OUTP + SZ_OUTP)             // ints: meta[16] | ptok[PSLOTS] | slot[NPAIR]

// Load 16 bf16 into the WMMA 16x32 per-lane layout:
// lane holds one row, K-chunks [c..c+7] and [c+16..c+23] (c = lane<16 ? 0 : 8).
__device__ __forceinline__ v16bf load_ab16(const bf16* p0, const bf16* p1) {
  v8bf lo = *(const v8bf*)p0;
  v8bf hi = *(const v8bf*)p1;
  return __builtin_shufflevector(lo, hi, 0, 1, 2, 3, 4, 5, 6, 7,
                                 8, 9, 10, 11, 12, 13, 14, 15);
}

#define WMMA_BF16(A, B, C) \
  __builtin_amdgcn_wmma_f32_16x16x32_bf16(false, (A), false, (B), (short)0, (C), false, false)

// ---------------- fp32 -> bf16 streaming convert ----------------
__global__ __launch_bounds__(256) void cvt_f32_bf16(const float* __restrict__ src,
                                                    bf16* __restrict__ dst, int n4) {
  int i = blockIdx.x * blockDim.x + threadIdx.x;
  if (i < n4) {
    float4 v = ((const float4*)src)[i];
    v4bf o = {(bf16)v.x, (bf16)v.y, (bf16)v.z, (bf16)v.w};
    ((v4bf*)dst)[i] = o;
  }
}

// ---------------- routing: expert histogram + 64-padded scatter ----------------
__global__ __launch_bounds__(256) void route_kernel(const int* __restrict__ topk_ids,
                                                    int* __restrict__ meta,
                                                    int* __restrict__ pair_token,
                                                    int* __restrict__ slot_of_pair) {
  __shared__ int cnt[E_];
  __shared__ int off[E_ + 1];
  __shared__ int cur[E_];
  if (threadIdx.x < E_) cnt[threadIdx.x] = 0;
  __syncthreads();
  for (int i = threadIdx.x; i < NPAIR; i += blockDim.x)
    atomicAdd(&cnt[topk_ids[i]], 1);
  __syncthreads();
  if (threadIdx.x == 0) {
    int o = 0;
    for (int e = 0; e < E_; ++e) { off[e] = o; o += (cnt[e] + 63) & ~63; }
    off[E_] = o;
  }
  __syncthreads();
  if (threadIdx.x < E_) cur[threadIdx.x] = off[threadIdx.x];
  if (threadIdx.x <= E_) meta[threadIdx.x] = off[threadIdx.x];
  for (int s = threadIdx.x; s < PSLOTS; s += blockDim.x) pair_token[s] = 0;  // pad rows -> token 0
  __syncthreads();
  for (int i = threadIdx.x; i < NPAIR; i += blockDim.x) {
    int e = topk_ids[i];
    int s = atomicAdd(&cur[e], 1);
    pair_token[s] = i / K_;
    slot_of_pair[i] = s;
  }
}

// ---------------- GEMM1 + SwiGLU: act[slot, I] = silu(h W1^T) * (h W3^T) ----------------
// Grid: (MTILES, I/128). Block: 256 = 8 waves as 2(M) x 4(N); block tile M=64, 128 gate+128 up cols.
// Wave: 2 M-tiles x 2 N-cols x (gate+up) -> 8 WMMA per k-step, 12 loads (1.5 loads/WMMA).
__global__ __launch_bounds__(256, 2) void moe_gemm1_swiglu(
    const bf16* __restrict__ hidb, const bf16* __restrict__ w13b,
    const int* __restrict__ meta, const int* __restrict__ pair_token,
    bf16* __restrict__ actb) {
  __shared__ int sOff[E_ + 1];
  if (threadIdx.x <= E_) sOff[threadIdx.x] = meta[threadIdx.x];
  __syncthreads();
  int row0 = blockIdx.x * 64;
  if (row0 >= sOff[E_]) return;
  int e = 0;
  while (row0 >= sOff[e + 1]) ++e;

  int lane = threadIdx.x & 31;
  int wv   = threadIdx.x >> 5;
  int mg   = wv >> 2;                       // 0..1  (M group of 32 rows)
  int ng   = wv & 3;                        // 0..3  (N group of 32 cols)
  int mr   = lane & 15;
  int koff = (lane < 16) ? 0 : 8;
  int rowm = row0 + mg * 32;
  int tok0 = pair_token[rowm + mr];
  int tok1 = pair_token[rowm + 16 + mr];
  int nb   = blockIdx.y * 128 + ng * 32;

  const bf16* ap0 = hidb + (size_t)tok0 * H_ + koff;
  const bf16* ap1 = hidb + (size_t)tok1 * H_ + koff;
  const bf16* bg0 = w13b + ((size_t)e * 2 * I_ + nb + mr) * H_ + koff;  // gate rows [0,I)
  const bf16* bg1 = bg0 + (size_t)16 * H_;
  const bf16* bu0 = bg0 + (size_t)I_ * H_;                              // up rows [I,2I)
  const bf16* bu1 = bu0 + (size_t)16 * H_;

  v8f cg00 = {}, cg01 = {}, cg10 = {}, cg11 = {};
  v8f cu00 = {}, cu01 = {}, cu10 = {}, cu11 = {};
#pragma unroll 2
  for (int k = 0; k < H_; k += 32) {
    __builtin_prefetch(bg0 + k + 512, 0, 1);   // global_prefetch_b8 on streamed weights
    __builtin_prefetch(bg1 + k + 512, 0, 1);
    __builtin_prefetch(bu0 + k + 512, 0, 1);
    __builtin_prefetch(bu1 + k + 512, 0, 1);
    v16bf a0 = load_ab16(ap0 + k, ap0 + k + 16);
    v16bf a1 = load_ab16(ap1 + k, ap1 + k + 16);
    v16bf g0 = load_ab16(bg0 + k, bg0 + k + 16);
    v16bf g1 = load_ab16(bg1 + k, bg1 + k + 16);
    v16bf u0 = load_ab16(bu0 + k, bu0 + k + 16);
    v16bf u1 = load_ab16(bu1 + k, bu1 + k + 16);
    cg00 = WMMA_BF16(a0, g0, cg00);
    cg01 = WMMA_BF16(a0, g1, cg01);
    cg10 = WMMA_BF16(a1, g0, cg10);
    cg11 = WMMA_BF16(a1, g1, cg11);
    cu00 = WMMA_BF16(a0, u0, cu00);
    cu01 = WMMA_BF16(a0, u1, cu01);
    cu10 = WMMA_BF16(a1, u0, cu10);
    cu11 = WMMA_BF16(a1, u1, cu11);
  }

  int mbase = (lane < 16) ? 0 : 8;                    // C layout: VGPR r -> row r (+8 hi lanes)
  size_t c00 = (size_t)(rowm + mbase) * I_ + nb + mr; // (m0, n0); n1 = +16; m1 = +16*I_
  size_t c10 = c00 + (size_t)16 * I_;
#pragma unroll
  for (int r = 0; r < 8; ++r) {
    size_t r0 = c00 + (size_t)r * I_;
    size_t r1 = c10 + (size_t)r * I_;
    float g, u;
    g = cg00[r]; u = cu00[r]; actb[r0]      = (bf16)(g / (1.0f + __expf(-g)) * u);
    g = cg01[r]; u = cu01[r]; actb[r0 + 16] = (bf16)(g / (1.0f + __expf(-g)) * u);
    g = cg10[r]; u = cu10[r]; actb[r1]      = (bf16)(g / (1.0f + __expf(-g)) * u);
    g = cg11[r]; u = cu11[r]; actb[r1 + 16] = (bf16)(g / (1.0f + __expf(-g)) * u);
  }
}

// ---------------- GEMM2: outp[slot, H] = act W2^T ----------------
// Grid: (MTILES, H/256). Block: 256 = 8 waves as 2(M) x 4(N); block tile M=64, N=256.
// Wave: 2 M-tiles x 4 N-tiles -> 8 WMMA per k-step, 12 loads (1.5 loads/WMMA).
__global__ __launch_bounds__(256, 2) void moe_gemm2(
    const bf16* __restrict__ actb, const bf16* __restrict__ w2b,
    const int* __restrict__ meta, float* __restrict__ outp) {
  __shared__ int sOff[E_ + 1];
  if (threadIdx.x <= E_) sOff[threadIdx.x] = meta[threadIdx.x];
  __syncthreads();
  int row0 = blockIdx.x * 64;
  if (row0 >= sOff[E_]) return;
  int e = 0;
  while (row0 >= sOff[e + 1]) ++e;

  int lane = threadIdx.x & 31;
  int wv   = threadIdx.x >> 5;
  int mg   = wv >> 2;                       // 0..1
  int ng   = wv & 3;                        // 0..3
  int mr   = lane & 15;
  int koff = (lane < 16) ? 0 : 8;
  int rowm = row0 + mg * 32;
  int nb   = blockIdx.y * 256 + ng * 64;

  const bf16* ap0 = actb + (size_t)(rowm + mr) * I_ + koff;
  const bf16* ap1 = ap0 + (size_t)16 * I_;
  const bf16* bp0 = w2b + ((size_t)e * H_ + nb + mr) * I_ + koff;
  const bf16* bp1 = bp0 + (size_t)16 * I_;
  const bf16* bp2 = bp0 + (size_t)32 * I_;
  const bf16* bp3 = bp0 + (size_t)48 * I_;

  v8f c00 = {}, c01 = {}, c02 = {}, c03 = {};
  v8f c10 = {}, c11 = {}, c12 = {}, c13 = {};
#pragma unroll 2
  for (int k = 0; k < I_; k += 32) {
    __builtin_prefetch(bp0 + k + 512, 0, 1);
    __builtin_prefetch(bp1 + k + 512, 0, 1);
    __builtin_prefetch(bp2 + k + 512, 0, 1);
    __builtin_prefetch(bp3 + k + 512, 0, 1);
    v16bf a0 = load_ab16(ap0 + k, ap0 + k + 16);
    v16bf a1 = load_ab16(ap1 + k, ap1 + k + 16);
    v16bf b0 = load_ab16(bp0 + k, bp0 + k + 16);
    v16bf b1 = load_ab16(bp1 + k, bp1 + k + 16);
    v16bf b2 = load_ab16(bp2 + k, bp2 + k + 16);
    v16bf b3 = load_ab16(bp3 + k, bp3 + k + 16);
    c00 = WMMA_BF16(a0, b0, c00);
    c01 = WMMA_BF16(a0, b1, c01);
    c02 = WMMA_BF16(a0, b2, c02);
    c03 = WMMA_BF16(a0, b3, c03);
    c10 = WMMA_BF16(a1, b0, c10);
    c11 = WMMA_BF16(a1, b1, c11);
    c12 = WMMA_BF16(a1, b2, c12);
    c13 = WMMA_BF16(a1, b3, c13);
  }

  int mbase = (lane < 16) ? 0 : 8;
  size_t cb0 = (size_t)(rowm + mbase) * H_ + nb + mr;  // (m0, n0); n+16/32/48; m1 = +16*H_
  size_t cb1 = cb0 + (size_t)16 * H_;
#pragma unroll
  for (int r = 0; r < 8; ++r) {
    size_t r0 = cb0 + (size_t)r * H_;
    size_t r1 = cb1 + (size_t)r * H_;
    outp[r0]      = c00[r];
    outp[r0 + 16] = c01[r];
    outp[r0 + 32] = c02[r];
    outp[r0 + 48] = c03[r];
    outp[r1]      = c10[r];
    outp[r1 + 16] = c11[r];
    outp[r1 + 32] = c12[r];
    outp[r1 + 48] = c13[r];
  }
}

// ---------------- combine: out[t] = w0 * outp[slot0] + w1 * outp[slot1] ----------------
__global__ __launch_bounds__(256) void combine_kernel(const float* __restrict__ outp,
                                                      const int* __restrict__ slot_of_pair,
                                                      const float* __restrict__ tw,
                                                      float* __restrict__ out) {
  int t = blockIdx.x;
  int s0 = slot_of_pair[t * K_ + 0];
  int s1 = slot_of_pair[t * K_ + 1];
  float w0 = tw[t * K_ + 0];
  float w1 = tw[t * K_ + 1];
  const float* p0 = outp + (size_t)s0 * H_;
  const float* p1 = outp + (size_t)s1 * H_;
  float* po = out + (size_t)t * H_;
  for (int h = threadIdx.x * 4; h < H_; h += blockDim.x * 4) {
    float4 a = *(const float4*)(p0 + h);
    float4 b = *(const float4*)(p1 + h);
    float4 r = {w0 * a.x + w1 * b.x, w0 * a.y + w1 * b.y,
                w0 * a.z + w1 * b.z, w0 * a.w + w1 * b.w};
    *(float4*)(po + h) = r;
  }
}

// ---------------- launch ----------------
extern "C" void kernel_launch(void* const* d_in, const int* in_sizes, int n_in,
                              void* d_out, int out_size, void* d_ws, size_t ws_size,
                              hipStream_t stream) {
  const float* hs  = (const float*)d_in[0];
  const float* tw  = (const float*)d_in[1];
  const int*   tid = (const int*)d_in[2];
  const float* w13 = (const float*)d_in[3];
  const float* w2  = (const float*)d_in[4];
  float* out = (float*)d_out;

  char* ws = (char*)d_ws;
  bf16*  w13b = (bf16*)(ws + OFF_W13B);
  bf16*  w2b  = (bf16*)(ws + OFF_W2B);
  bf16*  hidb = (bf16*)(ws + OFF_HIDB);
  bf16*  actb = (bf16*)(ws + OFF_ACTB);
  float* outp = (float*)(ws + OFF_OUTP);
  int*   meta = (int*)(ws + OFF_META);     // meta[0..E_] = padded expert offsets
  int*   ptok = meta + 16;                 // pair_token[PSLOTS]
  int*   slot = ptok + PSLOTS;             // slot_of_pair[NPAIR]

  // 1) bf16 conversion passes (weights then fit in 192MB L2 for all GEMM re-reads)
  int n4;
  n4 = (int)(SZ_W13B / 2 / 4);
  cvt_f32_bf16<<<(n4 + 255) / 256, 256, 0, stream>>>(w13, w13b, n4);
  n4 = (int)(SZ_W2B / 2 / 4);
  cvt_f32_bf16<<<(n4 + 255) / 256, 256, 0, stream>>>(w2, w2b, n4);
  n4 = (int)(SZ_HIDB / 2 / 4);
  cvt_f32_bf16<<<(n4 + 255) / 256, 256, 0, stream>>>(hs, hidb, n4);

  // 2) routing
  route_kernel<<<1, 256, 0, stream>>>(tid, meta, ptok, slot);

  // 3) grouped GEMM1 + SwiGLU (M=64 x 128 gate+up cols per block)
  moe_gemm1_swiglu<<<dim3(MTILES, I_ / 128), 256, 0, stream>>>(hidb, w13b, meta, ptok, actb);

  // 4) grouped GEMM2 (M=64 x N=256 per block)
  moe_gemm2<<<dim3(MTILES, H_ / 256), 256, 0, stream>>>(actb, w2b, meta, outp);

  // 5) weighted combine into output
  combine_kernel<<<T_, 256, 0, stream>>>(outp, slot, tw, out);
}
